// lstm_sender_agent_69114613728255
// MI455X (gfx1250) — matrix operand order repaired and checked by reading, not compiled, not run on hardware
//
#include <hip/hip_runtime.h>

// ---------------------------------------------------------------------------
// CDNA5 (gfx1250, wave32) bf16 WMMA implementation of the LSTM sender agent.
// All GEMMs use v_wmma_f32_16x16x32_bf16. Weights converted to bf16 N-major
// ([N][K]) once per call -> B operand is one contiguous 32B load per lane.
// Activations (h, x) live in bf16. Register tiling is sized to stay under
// ~200 VGPRs per wave (no scratch spills): hoisted A strips (128 VGPRs) for
// the K=512 GEMMs, 4x2 output blocking for the K=2048 GEMM.
// ---------------------------------------------------------------------------

typedef __attribute__((ext_vector_type(16))) __bf16 bf16x16;
typedef __attribute__((ext_vector_type(8)))  float  f32x8;

#define B_SZ    64
#define T_STEPS 40
#define H_SZ    512
#define E_SZ    512
#define V_SZ    32000
#define FH_SZ   512
#define FEAT_SZ 2048
#define NREG    49
#define MROWS   (B_SZ * NREG)   // 3136, multiple of 64

// ---------------- WMMA helpers ---------------------------------------------

__device__ __forceinline__ f32x8 wmma_bf16(bf16x16 a, bf16x16 b, f32x8 c) {
  return __builtin_amdgcn_wmma_f32_16x16x32_bf16(
      /*neg_a=*/false, a, /*neg_b=*/false, b,
      /*c_mod=*/(short)0, c, /*reuse_a=*/false, /*reuse_b=*/false);
}

// A-operand (16x32 bf16) from an f32 row-major matrix (cold path only).
__device__ __forceinline__ bf16x16 load_a_f32(const float* A, int m0, int k0, int ld) {
  int lane = threadIdx.x & 31;
  const float* p = A + (size_t)(m0 + (lane & 15)) * ld + k0 + ((lane >> 4) << 4);
  float4 f0 = *(const float4*)(p);
  float4 f1 = *(const float4*)(p + 4);
  float4 f2 = *(const float4*)(p + 8);
  float4 f3 = *(const float4*)(p + 12);
  bf16x16 o;
  o[0]  = (__bf16)f0.x; o[1]  = (__bf16)f0.y; o[2]  = (__bf16)f0.z; o[3]  = (__bf16)f0.w;
  o[4]  = (__bf16)f1.x; o[5]  = (__bf16)f1.y; o[6]  = (__bf16)f1.z; o[7]  = (__bf16)f1.w;
  o[8]  = (__bf16)f2.x; o[9]  = (__bf16)f2.y; o[10] = (__bf16)f2.z; o[11] = (__bf16)f2.w;
  o[12] = (__bf16)f3.x; o[13] = (__bf16)f3.y; o[14] = (__bf16)f3.z; o[15] = (__bf16)f3.w;
  return o;
}

// A-operand from a bf16 row-major matrix. Layout: lane m (m<16) holds
// row m, K = kb..kb+15 (contiguous, matches B), kb = (lane>>4)*16.
__device__ __forceinline__ bf16x16 load_a_bf16(const __bf16* A, int m0, int k0, int ld) {
  int lane = threadIdx.x & 31;
  const __bf16* p = A + (size_t)(m0 + (lane & 15)) * ld + k0 + ((lane >> 4) << 4);
  union { uint4 q[2]; bf16x16 v; } u;
  u.q[0] = *(const uint4*)(p);        // K = kb .. kb+7
  u.q[1] = *(const uint4*)(p + 8);    // K = kb+8 .. kb+15
  return u.v;
}

// B-operand (32x16 bf16) from an N-major [N][K] bf16 matrix: lane n (n<16)
// holds column n, K = k0..k0+15; lane n+16 holds K = k0+16..k0+31.
__device__ __forceinline__ bf16x16 load_b_nmaj(const __bf16* Bt, int n0, int k0, int ldk) {
  int lane = threadIdx.x & 31;
  const __bf16* p = Bt + (size_t)(n0 + (lane & 15)) * ldk + k0 + ((lane >> 4) << 4);
  return *(const bf16x16*)p;  // 32 contiguous bytes
}

// C/D f32 16x16 layout: VGPR i, lanes 0-15: M=i, N=lane; lanes 16-31: M=8+i.
__device__ __forceinline__ void store_c(float* C, f32x8 acc, int m0, int n0, int ldn) {
  int lane = threadIdx.x & 31;
  int n  = n0 + (lane & 15);
  int mb = m0 + ((lane >> 4) << 3);
#pragma unroll
  for (int i = 0; i < 8; ++i) C[(size_t)(mb + i) * ldn + n] = acc[i];
}

// ---------------- conversion / utility kernels -----------------------------

__global__ void cvt_bf16_kernel(const float* __restrict__ in, __bf16* __restrict__ out, int n) {
  int i = blockIdx.x * blockDim.x + threadIdx.x;
  if (i < n) out[i] = (__bf16)in[i];
}

// in f32 [R,C] row-major -> out bf16 [C,R] (N-major for GEMM B operand)
__global__ void cvt_transpose_kernel(const float* __restrict__ in, __bf16* __restrict__ out,
                                     int R, int C) {
  int i = blockIdx.x * blockDim.x + threadIdx.x;
  if (i >= R * C) return;
  int r = i / C, c = i - r * C;
  out[(size_t)c * R + r] = (__bf16)in[i];
}

__global__ void fill_bf16_kernel(__bf16* __restrict__ p, float v, int n) {
  int i = blockIdx.x * blockDim.x + threadIdx.x;
  if (i < n) p[i] = (__bf16)v;
}

// ---------------- GEMM kernels ---------------------------------------------

// f_emb = tanh(img @ W_fh + b) : A bf16 [3136,2048], Bt [512][2048], out f32.
// Each wave: 4 M-tiles x 2 N-tiles (acc 64 + a 32 + b 8 VGPRs, no spill);
// each B chunk feeds 4 WMMAs, each A chunk feeds 2.
__global__ void femb_gemm_kernel(const __bf16* __restrict__ A, const __bf16* __restrict__ Bt,
                                 const float* __restrict__ bias, float* __restrict__ C) {
  int w = (blockIdx.x * blockDim.x + threadIdx.x) >> 5;
  const int groupsN = (FH_SZ / 16) / 2;   // 16
  const int groupsM = MROWS / 64;         // 49
  int mg = w / groupsN, ng = w - mg * groupsN;
  if (mg >= groupsM) return;  // wave-uniform
  int m0b = mg * 64, n0b = ng * 32;
  f32x8 acc[4][2] = {};
  for (int k0 = 0; k0 < FEAT_SZ; k0 += 32) {
    bf16x16 a[4];
#pragma unroll
    for (int mi = 0; mi < 4; ++mi) a[mi] = load_a_bf16(A, m0b + mi * 16, k0, FEAT_SZ);
#pragma unroll
    for (int nj = 0; nj < 2; ++nj) {
      bf16x16 b = load_b_nmaj(Bt, n0b + nj * 16, k0, FEAT_SZ);
#pragma unroll
      for (int mi = 0; mi < 4; ++mi) acc[mi][nj] = wmma_bf16(a[mi], b, acc[mi][nj]);
    }
  }
  int lane = threadIdx.x & 31;
#pragma unroll
  for (int mi = 0; mi < 4; ++mi) {
    int mb = m0b + mi * 16 + ((lane >> 4) << 3);
#pragma unroll
    for (int nj = 0; nj < 2; ++nj) {
      int n = n0b + nj * 16 + (lane & 15);
      float bn = bias[n];
#pragma unroll
      for (int i = 0; i < 8; ++i)
        C[(size_t)(mb + i) * FH_SZ + n] = tanhf(acc[mi][nj][i] + bn);
    }
  }
}

// f_mean[b][j] = mean_n f_emb[b*49+n][j]
__global__ void mean_kernel(const float* __restrict__ femb, float* __restrict__ fmean) {
  int idx = blockIdx.x * blockDim.x + threadIdx.x;  // 64*512
  if (idx >= B_SZ * FH_SZ) return;
  int b = idx >> 9, j = idx & 511;
  float s = 0.f;
#pragma unroll 7
  for (int n = 0; n < NREG; ++n) s += femb[(size_t)(b * NREG + n) * FH_SZ + j];
  fmean[idx] = s * (1.0f / (float)NREG);
}

// out[64,512] = fmean(f32) @ Bt([512][512]) + bias.
// Writes f32 (c-state, Cb==null) or bf16 (h-state, Cf==null). Cold path.
__global__ void state_gemm_kernel(const float* __restrict__ A, const __bf16* __restrict__ Bt,
                                  const float* __restrict__ bias,
                                  float* __restrict__ Cf, __bf16* __restrict__ Cb) {
  int w = (blockIdx.x * blockDim.x + threadIdx.x) >> 5;
  const int tilesN = H_SZ / 16;  // 32
  int mt = w / tilesN, nt = w - mt * tilesN;
  if (mt >= B_SZ / 16) return;  // wave-uniform
  int m0 = mt * 16, n0 = nt * 16;
  f32x8 acc = {};
  for (int k0 = 0; k0 < FH_SZ; k0 += 32) {
    bf16x16 a = load_a_f32(A, m0, k0, FH_SZ);
    bf16x16 b = load_b_nmaj(Bt, n0, k0, FH_SZ);
    acc = wmma_bf16(a, b, acc);
  }
  int lane = threadIdx.x & 31;
  int n  = n0 + (lane & 15);
  int mb = m0 + ((lane >> 4) << 3);
  float bn = bias[n];
  if (Cf) {
#pragma unroll
    for (int i = 0; i < 8; ++i) Cf[(size_t)(mb + i) * H_SZ + n] = acc[i] + bn;
  } else {
#pragma unroll
    for (int i = 0; i < 8; ++i) Cb[(size_t)(mb + i) * H_SZ + n] = (__bf16)(acc[i] + bn);
  }
}

// gates[64,2048] = X @ WihT + Hprev @ WhhT + b  (K=512 each, X/H bf16).
// Each wave: 1 M-tile x 4 N-tiles; A strip (16x512 = 128 VGPRs) hoisted and
// reused across the 4 N-tiles, once for X, once for Hprev. ~170 VGPRs.
__global__ void gates_gemm_kernel(const __bf16* __restrict__ X, const __bf16* __restrict__ Hp,
                                  const __bf16* __restrict__ WihT, const __bf16* __restrict__ WhhT,
                                  const float* __restrict__ bias, float* __restrict__ G) {
  int w = (blockIdx.x * blockDim.x + threadIdx.x) >> 5;
  const int groupsN = (4 * H_SZ / 16) / 4;  // 32
  int mt = w / groupsN, ng = w - mt * groupsN;
  if (mt >= B_SZ / 16) return;  // wave-uniform
  int m0 = mt * 16, n0b = ng * 64;
  f32x8 acc[4] = {};
  bf16x16 a[16];
#pragma unroll
  for (int kk = 0; kk < 16; ++kk) a[kk] = load_a_bf16(X, m0, kk * 32, E_SZ);
#pragma unroll
  for (int j = 0; j < 4; ++j) {
    int n0 = n0b + j * 16;
#pragma unroll
    for (int kk = 0; kk < 16; ++kk)
      acc[j] = wmma_bf16(a[kk], load_b_nmaj(WihT, n0, kk * 32, E_SZ), acc[j]);
  }
#pragma unroll
  for (int kk = 0; kk < 16; ++kk) a[kk] = load_a_bf16(Hp, m0, kk * 32, H_SZ);
#pragma unroll
  for (int j = 0; j < 4; ++j) {
    int n0 = n0b + j * 16;
#pragma unroll
    for (int kk = 0; kk < 16; ++kk)
      acc[j] = wmma_bf16(a[kk], load_b_nmaj(WhhT, n0, kk * 32, H_SZ), acc[j]);
  }
  int lane = threadIdx.x & 31;
  int mb = m0 + ((lane >> 4) << 3);
#pragma unroll
  for (int j = 0; j < 4; ++j) {
    int n = n0b + j * 16 + (lane & 15);
    float bn = bias[n];
#pragma unroll
    for (int i = 0; i < 8; ++i)
      G[(size_t)(mb + i) * (4 * H_SZ) + n] = acc[j][i] + bn;
  }
}

// PyTorch gate order i,f,g,o; h written as bf16 (only ever a WMMA A-operand).
__global__ void lstm_cell_kernel(const float* __restrict__ G, __bf16* __restrict__ h,
                                 float* __restrict__ c) {
  int idx = blockIdx.x * blockDim.x + threadIdx.x;  // 64*512
  if (idx >= B_SZ * H_SZ) return;
  int b = idx >> 9, j = idx & 511;
  const float* g = G + (size_t)b * (4 * H_SZ);
  float gi = g[j], gf = g[j + 512], gg = g[j + 1024], go = g[j + 1536];
  float si = 1.f / (1.f + __expf(-gi));
  float sf = 1.f / (1.f + __expf(-gf));
  float so = 1.f / (1.f + __expf(-go));
  float cn = sf * c[idx] + si * tanhf(gg);
  c[idx] = cn;
  h[idx] = (__bf16)(so * tanhf(cn));
}

// logits[64,32000] = Htop(bf16,[64,512]) @ emb_bf16^T  (emb [V][E] = N-major).
// Each wave: 1 M-tile x 4 N-tiles; A strip hoisted into 128 VGPRs.
__global__ void logits_gemm_kernel(const __bf16* __restrict__ Htop,
                                   const __bf16* __restrict__ Emb, float* __restrict__ L) {
  int w = (blockIdx.x * blockDim.x + threadIdx.x) >> 5;
  const int groupsN = V_SZ / 64;  // 500
  int mt = w / groupsN, ng = w - mt * groupsN;
  if (mt >= B_SZ / 16) return;  // wave-uniform
  int m0 = mt * 16, n0b = ng * 64;
  bf16x16 a[16];
#pragma unroll
  for (int kk = 0; kk < 16; ++kk) a[kk] = load_a_bf16(Htop, m0, kk * 32, E_SZ);
  int lane = threadIdx.x & 31;
#pragma unroll
  for (int j = 0; j < 4; ++j) {
    int n0 = n0b + j * 16;
    const __bf16* bptr = Emb + (size_t)(n0 + (lane & 15)) * E_SZ + ((lane >> 4) << 4);
    if (j + 1 < 4)
      __builtin_prefetch((const void*)(bptr + 16 * E_SZ), 0, 1);  // next tile's rows
    f32x8 acc = {};
#pragma unroll
    for (int kk = 0; kk < 16; ++kk)
      acc = wmma_bf16(a[kk], *(const bf16x16*)(bptr + kk * 32), acc);
    store_c(L, acc, m0, n0, V_SZ);
  }
}

// Per-row: online logsumexp + Gumbel-max argmax; write token + log-prob.
__global__ void sample_kernel(const float* __restrict__ L, const float* __restrict__ g,
                              float* __restrict__ out_tok, float* __restrict__ out_lp,
                              int* __restrict__ tok_ws, int t) {
  int b = blockIdx.x;
  const float* row = L + (size_t)b * V_SZ;
  const float* gr  = g + (size_t)b * V_SZ;
  float m = -3.4e38f, s = 0.f, ybest = -3.4e38f;
  int vbest = 0;
  for (int v = threadIdx.x; v < V_SZ; v += blockDim.x) {
    float x = row[v];
    float nm = fmaxf(m, x);
    s = s * __expf(m - nm) + __expf(x - nm);
    m = nm;
    float y = x + gr[v];
    if (y > ybest) { ybest = y; vbest = v; }
  }
  __shared__ float sm[256], ss[256], sy[256];
  __shared__ int   sv[256];
  int tid = threadIdx.x;
  sm[tid] = m; ss[tid] = s; sy[tid] = ybest; sv[tid] = vbest;
  __syncthreads();
  for (int off = 128; off > 0; off >>= 1) {
    if (tid < off) {
      float m1 = sm[tid], s1 = ss[tid], m2 = sm[tid + off], s2 = ss[tid + off];
      float nm = fmaxf(m1, m2);
      ss[tid] = s1 * __expf(m1 - nm) + s2 * __expf(m2 - nm);
      sm[tid] = nm;
      float y2 = sy[tid + off];
      int   v2 = sv[tid + off];
      if (y2 > sy[tid] || (y2 == sy[tid] && v2 < sv[tid])) { sy[tid] = y2; sv[tid] = v2; }
    }
    __syncthreads();
  }
  if (tid == 0) {
    float lse = sm[0] + __logf(ss[0]);
    int tok = sv[0];
    tok_ws[b] = tok;
    out_tok[b * T_STEPS + t] = (float)tok;          // tokens.T  [B,T]
    out_lp[b * T_STEPS + t]  = row[tok] - lse;      // log_probs.T
  }
}

// x_next = emb_bf16[tok]  (stays bf16; feeds next step's gates A-operand)
__global__ void gather_emb_kernel(const __bf16* __restrict__ emb_bf, const int* __restrict__ tok,
                                  __bf16* __restrict__ xbuf) {
  int idx = blockIdx.x * blockDim.x + threadIdx.x;  // 64*512
  if (idx >= B_SZ * E_SZ) return;
  int b = idx >> 9, e = idx & 511;
  xbuf[idx] = emb_bf[(size_t)tok[b] * E_SZ + e];
}

// ---------------- launch ----------------------------------------------------

extern "C" void kernel_launch(void* const* d_in, const int* in_sizes, int n_in,
                              void* d_out, int out_size, void* d_ws, size_t ws_size,
                              hipStream_t stream) {
  (void)in_sizes; (void)n_in; (void)out_size; (void)ws_size;

  const float* img    = (const float*)d_in[0];   // [64,49,2048]
  const float* W_fh   = (const float*)d_in[1];   // [2048,512]
  const float* b_fh   = (const float*)d_in[2];   // [512]
  const float* Wc     = (const float*)d_in[3];   // [2,512,512]
  const float* bc     = (const float*)d_in[4];   // [2,512]
  const float* Wh     = (const float*)d_in[5];   // [2,512,512]
  const float* bh     = (const float*)d_in[6];   // [2,512]
  const float* W_ih   = (const float*)d_in[7];   // [2,512,2048]
  const float* W_hh   = (const float*)d_in[8];   // [2,512,2048]
  const float* b_lstm = (const float*)d_in[9];   // [2,2048]
  const float* emb    = (const float*)d_in[10];  // [32000,512]
  const float* gumbel = (const float*)d_in[11];  // [40,64,32000]
  float* out = (float*)d_out;                    // tokens.T (2560) ++ log_probs.T (2560)

  // ---- workspace carve (256B aligned) ----
  char* p = (char*)d_ws;
  auto carve = [&](size_t bytes) -> void* {
    void* r = (void*)p;
    p += (bytes + 255) & ~(size_t)255;
    return r;
  };
  __bf16* img_bf = (__bf16*)carve((size_t)MROWS * FEAT_SZ * 2);
  __bf16* WfhT   = (__bf16*)carve((size_t)FH_SZ * FEAT_SZ * 2);     // [512][2048]
  __bf16* WcT    = (__bf16*)carve((size_t)2 * H_SZ * FH_SZ * 2);    // [l][512][512]
  __bf16* WhT    = (__bf16*)carve((size_t)2 * H_SZ * FH_SZ * 2);
  __bf16* WihT   = (__bf16*)carve((size_t)2 * 4 * H_SZ * E_SZ * 2); // [l][2048][512]
  __bf16* WhhT   = (__bf16*)carve((size_t)2 * 4 * H_SZ * H_SZ * 2);
  __bf16* emb_bf = (__bf16*)carve((size_t)V_SZ * E_SZ * 2);
  float*  femb   = (float*)carve((size_t)MROWS * FH_SZ * 4);
  float*  fmean  = (float*)carve((size_t)B_SZ * FH_SZ * 4);
  __bf16* hbuf   = (__bf16*)carve((size_t)2 * B_SZ * H_SZ * 2);     // bf16 state
  float*  cbuf   = (float*)carve((size_t)2 * B_SZ * H_SZ * 4);
  float*  gates  = (float*)carve((size_t)B_SZ * 4 * H_SZ * 4);
  float*  logits = (float*)carve((size_t)B_SZ * V_SZ * 4);
  __bf16* xbuf   = (__bf16*)carve((size_t)B_SZ * E_SZ * 2);
  int*    tok    = (int*)carve((size_t)B_SZ * 4);

  const dim3 blk(256);
  auto nb = [](int n) { return dim3((unsigned)((n + 255) / 256)); };

  // ---- one-time (per call) weight conversion / transposition ----
  cvt_bf16_kernel<<<nb(MROWS * FEAT_SZ), blk, 0, stream>>>(img, img_bf, MROWS * FEAT_SZ);
  cvt_bf16_kernel<<<nb(V_SZ * E_SZ), blk, 0, stream>>>(emb, emb_bf, V_SZ * E_SZ);
  cvt_transpose_kernel<<<nb(FEAT_SZ * FH_SZ), blk, 0, stream>>>(W_fh, WfhT, FEAT_SZ, FH_SZ);
  for (int l = 0; l < 2; ++l) {
    cvt_transpose_kernel<<<nb(FH_SZ * H_SZ), blk, 0, stream>>>(
        Wc + (size_t)l * FH_SZ * H_SZ, WcT + (size_t)l * H_SZ * FH_SZ, FH_SZ, H_SZ);
    cvt_transpose_kernel<<<nb(FH_SZ * H_SZ), blk, 0, stream>>>(
        Wh + (size_t)l * FH_SZ * H_SZ, WhT + (size_t)l * H_SZ * FH_SZ, FH_SZ, H_SZ);
    cvt_transpose_kernel<<<nb(E_SZ * 4 * H_SZ), blk, 0, stream>>>(
        W_ih + (size_t)l * E_SZ * 4 * H_SZ, WihT + (size_t)l * 4 * H_SZ * E_SZ, E_SZ, 4 * H_SZ);
    cvt_transpose_kernel<<<nb(H_SZ * 4 * H_SZ), blk, 0, stream>>>(
        W_hh + (size_t)l * H_SZ * 4 * H_SZ, WhhT + (size_t)l * 4 * H_SZ * H_SZ, H_SZ, 4 * H_SZ);
  }

  // ---- feature embedding + initial state ----
  // waves = 49 M-groups x 16 N-groups = 784 -> 98 blocks of 8 waves
  femb_gemm_kernel<<<dim3(98), blk, 0, stream>>>(img_bf, WfhT, b_fh, femb);
  mean_kernel<<<nb(B_SZ * FH_SZ), blk, 0, stream>>>(femb, fmean);
  for (int l = 0; l < 2; ++l) {
    state_gemm_kernel<<<dim3(16), blk, 0, stream>>>(
        fmean, WcT + (size_t)l * H_SZ * FH_SZ, bc + l * H_SZ,
        cbuf + (size_t)l * B_SZ * H_SZ, (__bf16*)nullptr);       // c0: f32
    state_gemm_kernel<<<dim3(16), blk, 0, stream>>>(
        fmean, WhT + (size_t)l * H_SZ * FH_SZ, bh + l * H_SZ,
        (float*)nullptr, hbuf + (size_t)l * B_SZ * H_SZ);        // h0: bf16
  }
  fill_bf16_kernel<<<nb(B_SZ * E_SZ), blk, 0, stream>>>(xbuf, 3.0f, B_SZ * E_SZ);  // START_IDX

  // ---- autoregressive loop ----
  for (int t = 0; t < T_STEPS; ++t) {
    // layer 0: input = xbuf, recurrent = h[0]   (128 waves -> 16 blocks)
    gates_gemm_kernel<<<dim3(16), blk, 0, stream>>>(
        xbuf, hbuf, WihT, WhhT, b_lstm, gates);
    lstm_cell_kernel<<<nb(B_SZ * H_SZ), blk, 0, stream>>>(gates, hbuf, cbuf);
    // layer 1: input = updated h[0], recurrent = h[1]
    gates_gemm_kernel<<<dim3(16), blk, 0, stream>>>(
        hbuf, hbuf + B_SZ * H_SZ,
        WihT + (size_t)4 * H_SZ * E_SZ, WhhT + (size_t)4 * H_SZ * H_SZ,
        b_lstm + 4 * H_SZ, gates);
    lstm_cell_kernel<<<nb(B_SZ * H_SZ), blk, 0, stream>>>(
        gates, hbuf + B_SZ * H_SZ, cbuf + B_SZ * H_SZ);
    // logits + sample + next input (2000 waves -> 250 blocks)
    logits_gemm_kernel<<<dim3(250), blk, 0, stream>>>(
        hbuf + B_SZ * H_SZ, emb_bf, logits);
    sample_kernel<<<dim3(B_SZ), blk, 0, stream>>>(
        logits, gumbel + (size_t)t * B_SZ * V_SZ, out, out + B_SZ * T_STEPS, tok, t);
    gather_emb_kernel<<<nb(B_SZ * E_SZ), blk, 0, stream>>>(emb_bf, tok, xbuf);
  }
}
